// SetTransformer_82308753261088
// MI455X (gfx1250) — compile-verified, matrix operand order checked by reading, not run on hardware
//
#include <hip/hip_runtime.h>
#include <hip/hip_bf16.h>

// ---------------------------------------------------------------------------
// Set Transformer forward for MI455X (gfx1250, wave32, WMMA).
// All dense GEMMs use v_wmma_f32_16x16x32_bf16 (f32 data converted to bf16,
// staged in LDS in a fragment-contiguous swizzled layout, f32 accumulation).
// Softmax / layernorm / gelu epilogues stay in f32.
// ---------------------------------------------------------------------------

#define D_MODEL 512
#define H_DIM   2048
#define N_HEADS 8
#define HEAD_D  64
#define N_INDS  64
#define N_OUT   10
#define BATCH   8
#define S_ENC   (N_INDS + 512)   // 576
#define S_DEC   (N_OUT + N_INDS) // 74

typedef __attribute__((ext_vector_type(16))) __bf16 v16bf;
typedef __attribute__((ext_vector_type(8)))  float  v8f;

union Frag16 { v16bf v; uint4 q[2]; };

__device__ __forceinline__ unsigned int f32_to_bf16_bits(float f) {
  unsigned int x = __float_as_uint(f);
  return (x + 0x7FFFu + ((x >> 16) & 1u)) >> 16;   // round-to-nearest-even
}

__device__ __forceinline__ unsigned int pack_bf16x2(float lo, float hi) {
  return f32_to_bf16_bits(lo) | (f32_to_bf16_bits(hi) << 16);
}

__device__ __forceinline__ float geluf(float x) {
  return 0.5f * x * (1.0f + erff(x * 0.70710678118654752f));
}

// ---------------------------------------------------------------------------
// Generic GEMM: C[M,N] = act(A[M,K] @ W[K,N] + bias)
// Row-major, lda=K, ldb=N, ldc=N. K % 32 == 0, N % 64 == 0, M arbitrary.
// Block = 256 threads (8 waves). Tile 64x64, K-step 32.
// Waves laid out 4(M) x 2(N); each wave -> two 16x16 WMMA accumulators.
//
// LDS tiles are stored fragment-contiguous: each row (m for A, n for B) holds
// its 32 bf16 K-values in block order {k0-7, k16-23, k8-15, k24-31} so that
// lanes 0-15 fetch halves [k0..7][k16..23] and lanes 16-31 fetch
// [k8..15][k24..31] as two contiguous ds_load_b128 — exactly the ISA 16-bit
// A/B operand layout for v_wmma_*_16x16x32.
// Row stride = 20 dwords (80 B): 16-byte aligned, conflict-free mod 64 banks.
// ---------------------------------------------------------------------------
__global__ __launch_bounds__(256) void gemm_bf16_wmma(
    const float* __restrict__ A, const float* __restrict__ W,
    const float* __restrict__ bias, float* __restrict__ C,
    int M, int N, int K, int act)
{
  __shared__ unsigned int As[64][20];   // A tile, m-major, swizzled K blocks
  __shared__ unsigned int Bs[64][20];   // B tile, n-major, swizzled K blocks

  const int tid  = threadIdx.x;
  const int lane = tid & 31;
  const int wave = tid >> 5;
  const int wm   = wave >> 1;               // 0..3 : M sub-tile
  const int wn   = wave & 1;                // 0..1 : N sub-tile
  const int lm   = lane & 15;
  const int fo   = (lane < 16) ? 0 : 8;     // dword offset into fragment row
  const int m0   = blockIdx.y * 64;
  const int n0   = blockIdx.x * 64;

  // A cooperative-load coords: thread -> (row, k-group of 8)
  const int arow  = tid >> 2;               // 0..63
  const int akg   = tid & 3;                // 0..3
  const int abidx = ((akg & 1) << 1) | (akg >> 1);   // k-block swizzle
  // B cooperative-load coords: thread -> (col n, k-group of 8)
  const int bn    = tid & 63;               // 0..63
  const int bkg   = tid >> 6;               // 0..3
  const int bbidx = ((bkg & 1) << 1) | (bkg >> 1);

  v8f acc0 = {};
  v8f acc1 = {};

  for (int k0 = 0; k0 < K; k0 += 32) {
    { // ---- A tile 64x32: two global b128 loads, branchless row masking ----
      const int gr = m0 + arow;
      const int sr = (gr < M) ? gr : (M - 1);          // clamped, always valid
      const float* ap = A + (size_t)sr * K + k0 + akg * 8;
      float4 f0 = *(const float4*)(ap);
      float4 f1 = *(const float4*)(ap + 4);
      if (gr >= M) {                                    // lowers to v_cndmask
        f0.x = 0.f; f0.y = 0.f; f0.z = 0.f; f0.w = 0.f;
        f1.x = 0.f; f1.y = 0.f; f1.z = 0.f; f1.w = 0.f;
      }
      uint4 pk;
      pk.x = pack_bf16x2(f0.x, f0.y);
      pk.y = pack_bf16x2(f0.z, f0.w);
      pk.z = pack_bf16x2(f1.x, f1.y);
      pk.w = pack_bf16x2(f1.z, f1.w);
      *(uint4*)&As[arow][abidx * 4] = pk;               // ds_store_b128
    }
    { // ---- B tile 32x64, stored n-major: 8 coalesced b32 loads ------------
      const float* bp = W + (size_t)(k0 + bkg * 8) * N + n0 + bn;
      float f[8];
      #pragma unroll
      for (int i = 0; i < 8; ++i) f[i] = bp[(size_t)i * N];
      uint4 pk;
      pk.x = pack_bf16x2(f[0], f[1]);
      pk.y = pack_bf16x2(f[2], f[3]);
      pk.z = pack_bf16x2(f[4], f[5]);
      pk.w = pack_bf16x2(f[6], f[7]);
      *(uint4*)&Bs[bn][bbidx * 4] = pk;                 // ds_store_b128
    }
    __syncthreads();

    // ---- fragments: 2x ds_load_b128 each, already in operand order --------
    Frag16 af, b0f, b1f;
    const unsigned int* ar = &As[wm * 16 + lm][0];
    af.q[0]  = *(const uint4*)(ar + fo);
    af.q[1]  = *(const uint4*)(ar + fo + 4);
    const unsigned int* b0r = &Bs[wn * 32 + lm][0];
    b0f.q[0] = *(const uint4*)(b0r + fo);
    b0f.q[1] = *(const uint4*)(b0r + fo + 4);
    const unsigned int* b1r = &Bs[wn * 32 + 16 + lm][0];
    b1f.q[0] = *(const uint4*)(b1r + fo);
    b1f.q[1] = *(const uint4*)(b1r + fo + 4);

    acc0 = __builtin_amdgcn_wmma_f32_16x16x32_bf16(false, af.v, false, b0f.v,
                                                   (short)0, acc0, false, false);
    acc1 = __builtin_amdgcn_wmma_f32_16x16x32_bf16(false, af.v, false, b1f.v,
                                                   (short)0, acc1, false, false);
    __syncthreads();
  }

  // epilogue: C/D layout — VGPR r: lanes 0-15 -> M=r, lanes 16-31 -> M=r+8
  const int rbase = (lane < 16) ? 0 : 8;
  const int c0 = n0 + wn * 32 + lm;
  const int c1 = c0 + 16;
  const float bias0 = bias ? bias[c0] : 0.0f;
  const float bias1 = bias ? bias[c1] : 0.0f;
  #pragma unroll
  for (int r = 0; r < 8; ++r) {
    const int row = m0 + wm * 16 + rbase + r;
    if (row < M) {
      float v0 = acc0[r] + bias0;
      float v1 = acc1[r] + bias1;
      if (act) { v0 = geluf(v0); v1 = geluf(v1); }
      C[(size_t)row * N + c0] = v0;
      C[(size_t)row * N + c1] = v1;
    }
  }
}

// ---------------------------------------------------------------------------
// Per-row online-softmax attention. grid.x = BATCH * N_HEADS.
// q/k/v/ctx layout: (B, S, 512); head h occupies cols [h*64, h*64+64).
// ---------------------------------------------------------------------------
__global__ __launch_bounds__(256) void attention_kernel(
    const float* __restrict__ q, const float* __restrict__ k,
    const float* __restrict__ v, float* __restrict__ ctx, int S)
{
  const int b = blockIdx.x >> 3;
  const int h = blockIdx.x & 7;
  const size_t base = (size_t)b * S * 512 + (size_t)h * 64;

  for (int row = threadIdx.x; row < S; row += blockDim.x) {
    const float* qr_p = q + base + (size_t)row * 512;
    float qr[64];
    #pragma unroll
    for (int d = 0; d < 64; ++d) qr[d] = qr_p[d];

    float mx = -3.0e38f, l = 0.0f;
    float acc[64];
    #pragma unroll
    for (int d = 0; d < 64; ++d) acc[d] = 0.0f;

    for (int j = 0; j < S; ++j) {
      const float* kr = k + base + (size_t)j * 512;
      float s = 0.0f;
      #pragma unroll
      for (int d = 0; d < 64; ++d) s += qr[d] * kr[d];
      s *= 0.125f;                       // 1/sqrt(64)
      const float mn = fmaxf(mx, s);
      const float sc = __expf(mx - mn);
      const float p  = __expf(s - mn);
      l = l * sc + p;
      const float* vr = v + base + (size_t)j * 512;
      #pragma unroll
      for (int d = 0; d < 64; ++d) acc[d] = acc[d] * sc + p * vr[d];
      mx = mn;
    }
    const float inv = 1.0f / l;
    float* cr = ctx + base + (size_t)row * 512;
    #pragma unroll
    for (int d = 0; d < 64; ++d) cr[d] = acc[d] * inv;
  }
}

// ---------------------------------------------------------------------------
// In-place layernorm over rows of width 512. grid.x = #rows, 256 threads.
// ---------------------------------------------------------------------------
__global__ __launch_bounds__(256) void layernorm_rows(
    float* __restrict__ x, const float* __restrict__ g, const float* __restrict__ b)
{
  __shared__ float red[256];
  const int tid = threadIdx.x;
  float* xr = x + (size_t)blockIdx.x * 512;

  const float v0 = xr[tid];
  const float v1 = xr[tid + 256];

  red[tid] = v0 + v1;
  __syncthreads();
  for (int off = 128; off > 0; off >>= 1) {
    if (tid < off) red[tid] += red[tid + off];
    __syncthreads();
  }
  const float mean = red[0] * (1.0f / 512.0f);
  __syncthreads();

  const float d0 = v0 - mean, d1 = v1 - mean;
  red[tid] = d0 * d0 + d1 * d1;
  __syncthreads();
  for (int off = 128; off > 0; off >>= 1) {
    if (tid < off) red[tid] += red[tid + off];
    __syncthreads();
  }
  const float var = red[0] * (1.0f / 512.0f);
  const float rstd = rsqrtf(var + 1e-5f);

  xr[tid]       = d0 * rstd * g[tid]       + b[tid];
  xr[tid + 256] = d1 * rstd * g[tid + 256] + b[tid + 256];
}

// ---------------------------------------------------------------------------
// Glue kernels
// ---------------------------------------------------------------------------
__global__ void build_enc_input(const float* __restrict__ inductor,
                                const float* __restrict__ followee,
                                float* __restrict__ x, int total)
{
  int idx = blockIdx.x * blockDim.x + threadIdx.x;
  if (idx >= total) return;
  const int c = idx & 511;
  const int r = (idx >> 9) % S_ENC;
  const int b = idx / (S_ENC * 512);
  x[idx] = (r < N_INDS)
         ? inductor[r * 512 + c]
         : followee[((size_t)b * 512 + (r - N_INDS)) * 512 + c];
}

__global__ void build_dec_input(const float* __restrict__ seed,
                                const float* __restrict__ encoded,  // (8, 576, 512)
                                float* __restrict__ x, int total)
{
  int idx = blockIdx.x * blockDim.x + threadIdx.x;
  if (idx >= total) return;
  const int c = idx & 511;
  const int r = (idx >> 9) % S_DEC;
  const int b = idx / (S_DEC * 512);
  x[idx] = (r < N_OUT)
         ? seed[r * 512 + c]
         : encoded[((size_t)b * S_ENC + (r - N_OUT)) * 512 + c];
}

__global__ void extract_query(const float* __restrict__ decoded,  // (8, 74, 512)
                              float* __restrict__ qy, int total)   // (80, 512)
{
  int idx = blockIdx.x * blockDim.x + threadIdx.x;
  if (idx >= total) return;
  const int c = idx & 511;
  const int o = (idx >> 9) % N_OUT;
  const int b = idx / (N_OUT * 512);
  qy[idx] = decoded[((size_t)b * S_DEC + o) * 512 + c];
}

// ---------------------------------------------------------------------------
// Scoring: out[b,o,c] = sum_j gelu(hq[b*10+o,j] + hc[b*512+c,j] + b1[j])*w2[j] + b2
// grid = (512/8, 80); block 256 = 8 waves; one wave per candidate.
// ---------------------------------------------------------------------------
__global__ __launch_bounds__(256) void score_kernel(
    const float* __restrict__ hq,  // (80, 2048)
    const float* __restrict__ hc,  // (4096, 2048)
    const float* __restrict__ b1, const float* __restrict__ w2,
    const float* __restrict__ b2, float* __restrict__ out)
{
  __shared__ float hqs[H_DIM];
  const int bo = blockIdx.y;            // b*10 + o
  const int b  = bo / N_OUT;
  for (int j = threadIdx.x; j < H_DIM; j += 256) hqs[j] = hq[(size_t)bo * H_DIM + j];
  __syncthreads();

  const int wave = threadIdx.x >> 5;
  const int lane = threadIdx.x & 31;
  const int c = blockIdx.x * 8 + wave;
  const float* hcrow = hc + ((size_t)b * 512 + c) * H_DIM;

  float s = 0.0f;
  for (int j = lane; j < H_DIM; j += 32) {
    const float x = hqs[j] + hcrow[j] + b1[j];
    s += geluf(x) * w2[j];
  }
  #pragma unroll
  for (int off = 16; off > 0; off >>= 1) s += __shfl_xor(s, off, 32);
  if (lane == 0) out[(size_t)bo * 512 + c] = s + b2[0];
}

// ---------------------------------------------------------------------------
// Host side
// ---------------------------------------------------------------------------
static void launch_gemm(const float* A, const float* W, const float* bias,
                        float* C, int M, int N, int K, int act, hipStream_t s)
{
  dim3 grid(N / 64, (M + 63) / 64);
  gemm_bf16_wmma<<<grid, 256, 0, s>>>(A, W, bias, C, M, N, K, act);
}

// p[]: 0 wq,1 bq,2 wk,3 bk,4 wv,5 bv,6 wo,7 bo,8 ln1g,9 ln1b,10 w1,11 b1,
//      12 w2,13 b2,14 ln2g,15 ln2b
static void run_block(const float* x, int S, const float* const* p,
                      float* q, float* k, float* v, float* ctx, float* a,
                      float* h, float* o, hipStream_t s)
{
  const int M = BATCH * S;
  launch_gemm(x, p[0], p[1], q, M, 512, 512, 0, s);
  launch_gemm(x, p[2], p[3], k, M, 512, 512, 0, s);
  launch_gemm(x, p[4], p[5], v, M, 512, 512, 0, s);
  attention_kernel<<<BATCH * N_HEADS, 256, 0, s>>>(q, k, v, ctx, S);
  launch_gemm(ctx, p[6], p[7], a, M, 512, 512, 0, s);
  layernorm_rows<<<M, 256, 0, s>>>(a, p[8], p[9]);
  launch_gemm(a, p[10], p[11], h, M, H_DIM, 512, 1, s);     // gelu
  launch_gemm(h, p[12], p[13], o, M, 512, H_DIM, 0, s);
  layernorm_rows<<<M, 256, 0, s>>>(o, p[14], p[15]);
}

extern "C" void kernel_launch(void* const* d_in, const int* in_sizes, int n_in,
                              void* d_out, int out_size, void* d_ws, size_t ws_size,
                              hipStream_t stream)
{
  (void)in_sizes; (void)n_in; (void)out_size; (void)ws_size;

  // setup_inputs() leaves, depth-first in dict insertion order:
  const float* followee = (const float*)d_in[0];   // (8, 512, 512)
  const float* cand     = (const float*)d_in[1];   // (8, 512, 512)
  const float* inductor = (const float*)d_in[2];   // (1, 64, 512)
  const float* dseed    = (const float*)d_in[3];   // (1, 10, 512)
  const float* encp[16];
  const float* decp[16];
  for (int i = 0; i < 16; ++i) {
    encp[i] = (const float*)d_in[4 + i];
    decp[i] = (const float*)d_in[20 + i];
  }
  const float* score_wq = (const float*)d_in[36];  // (512, 2048)
  const float* score_wc = (const float*)d_in[37];  // (512, 2048)
  const float* score_b1 = (const float*)d_in[38];  // (2048)
  const float* score_w2 = (const float*)d_in[39];  // (2048, 1)
  const float* score_b2 = (const float*)d_in[40];  // (1)
  float* out = (float*)d_out;                      // (8, 10, 512)

  // Workspace layout (floats): five (4608*512) buffers + one (4608*2048).
  const size_t SZ = (size_t)BATCH * S_ENC * D_MODEL;  // 2,359,296
  float* B0 = (float*)d_ws;
  float* B1 = B0 + SZ;
  float* B2 = B1 + SZ;
  float* B3 = B2 + SZ;
  float* B4 = B3 + SZ;
  float* BH = B4 + SZ;                               // 4608 * 2048

  // ---- encoder -----------------------------------------------------------
  {
    const int total = BATCH * S_ENC * D_MODEL;
    build_enc_input<<<(total + 255) / 256, 256, 0, stream>>>(inductor, followee, B0, total);
  }
  // x=B0 q=B1 k=B2 v=B3 ctx=B4 a=B0(x dead) h=BH o=B4(ctx dead) -> encoded=B4
  run_block(B0, S_ENC, encp, B1, B2, B3, B4, B0, BH, B4, stream);

  // ---- decoder -----------------------------------------------------------
  {
    const int total = BATCH * S_DEC * D_MODEL;
    build_dec_input<<<(total + 255) / 256, 256, 0, stream>>>(dseed, B4, B0, total);
  }
  // x=B0 q=B1 k=B2 v=B3 ctx=B4(encoded dead) a=B0 h=BH o=B1(q dead) -> decoded=B1
  run_block(B0, S_DEC, decp, B1, B2, B3, B4, B0, BH, B1, stream);

  // ---- scoring head ------------------------------------------------------
  {
    const int total = BATCH * N_OUT * D_MODEL;       // 40960
    extract_query<<<(total + 255) / 256, 256, 0, stream>>>(B1, B4, total);
  }
  launch_gemm(B4,   score_wq, nullptr, B3, BATCH * N_OUT, H_DIM, 512, 0, stream); // hq (80,2048)
  launch_gemm(cand, score_wc, nullptr, BH, BATCH * 512,   H_DIM, 512, 0, stream); // hc (4096,2048)

  {
    dim3 grid(512 / 8, BATCH * N_OUT);
    score_kernel<<<grid, 256, 0, stream>>>(B3, BH, score_b1, score_w2, score_b2, out);
  }
}
